// GraphTransformer_89790586290225
// MI455X (gfx1250) — matrix operand order repaired and checked by reading, not compiled, not run on hardware
//
#include <hip/hip_runtime.h>
#include <hip/hip_bf16.h>
#include <cstdint>

#define N_NODES 20000
#define N_EDGES 160000
#define DMODEL  128
#define NHEAD   8
#define CHEAD   128
#define HC      1024   // NHEAD*CHEAD
#define NLAYER  4
#define BGRAPH  32
#define FFDIM   512

typedef __attribute__((ext_vector_type(16))) _Float16 v16h;
typedef __attribute__((ext_vector_type(8)))  _Float16 v8h;
typedef __attribute__((ext_vector_type(4)))  _Float16 v4h;
typedef __attribute__((ext_vector_type(8)))  float    v8f;

// ---------- monotonic float<->uint mapping for atomic segment-max ----------
__device__ __forceinline__ unsigned fmap(float f) {
    unsigned u = __float_as_uint(f);
    return (u & 0x80000000u) ? ~u : (u | 0x80000000u);
}
__device__ __forceinline__ float funmap(unsigned u) {
    unsigned b = (u & 0x80000000u) ? (u & 0x7FFFFFFFu) : ~u;
    return __uint_as_float(b);
}

// ---------- h = x @ in_w + in_b + PE[time_steps] ----------
__global__ void init_h_kernel(const float* __restrict__ x, const int* __restrict__ ts,
                              const float* __restrict__ in_w, const float* __restrict__ in_b,
                              float* __restrict__ h, _Float16* __restrict__ h16) {
    int tid = blockIdx.x * blockDim.x + threadIdx.x;
    if (tid >= N_NODES * DMODEL) return;
    int n = tid >> 7, c = tid & 127;
    float acc = in_b[c]
              + x[n * 3 + 0] * in_w[0 * DMODEL + c]
              + x[n * 3 + 1] * in_w[1 * DMODEL + c]
              + x[n * 3 + 2] * in_w[2 * DMODEL + c];
    int p = c & ~1;  // even pair index
    float freq = __expf(-(float)p * (9.210340371976184f / 128.0f));  // ln(10000)/D
    float ang = (float)ts[n] * freq;
    acc += (c & 1) ? __cosf(ang) : __sinf(ang);
    h[tid] = acc;
    h16[tid] = (_Float16)acc;
}

// ---------- weight pack: W[K][N] f32 (row-major) -> Wt[N][K] f16 ----------
__global__ void pack_wt_kernel(const float* __restrict__ W, _Float16* __restrict__ Wt,
                               int K, int N) {
    int tid = blockIdx.x * blockDim.x + threadIdx.x;
    if (tid >= K * N) return;
    int n = tid / K, kk = tid - n * K;
    Wt[tid] = (_Float16)W[(size_t)kk * N + n];
}

// ---------- WMMA GEMM: out[M,N] = A[M,K](f16) x Bt[N,K](f16)^T + bias ----------
// one wave computes a 32(M) x 64(N) tile (2 A fragments x 4 B tiles, B reused);
// block = 4 waves; requires M % 32 == 0, K % 32 == 0, N % 64 == 0
__global__ void wmma_gemm_kernel(const _Float16* __restrict__ A,
                                 const _Float16* __restrict__ Bt,
                                 const float* __restrict__ bias,
                                 float* __restrict__ outF,
                                 _Float16* __restrict__ outH,
                                 int M, int K, int N, int relu) {
    const int lane = threadIdx.x & 31;
    const int wave = threadIdx.x >> 5;
    const int m0 = blockIdx.x << 5;                  // 32 rows per wave-tile
    const int n0 = (blockIdx.y * 4 + wave) << 6;     // 64 cols per wave
    if (n0 >= N) return;                 // wave-uniform guard (EXEC all-ones at WMMA)

    const int hi = lane >> 4;            // half-wave select
    const int mr = lane & 15;            // row (A) / column (B,C,D) within tile

    v8f acc0[4] = {v8f{}, v8f{}, v8f{}, v8f{}};
    v8f acc1[4] = {v8f{}, v8f{}, v8f{}, v8f{}};
    const _Float16* arow0 = A + (size_t)(m0 + mr) * K;
    const _Float16* arow1 = arow0 + (size_t)16 * K;

    for (int kc = 0; kc < K; kc += 32) {
        // A 16x32 fragments: lanes0-15 K 0..7 / 16..23 ; lanes16-31 K 8..15 / 24..31
        union { v16h v; v8h h[2]; } a0, a1;
        a0.h[0] = *(const v8h*)(arow0 + kc + hi * 8);
        a0.h[1] = *(const v8h*)(arow0 + kc + 16 + hi * 8);
        a1.h[0] = *(const v8h*)(arow1 + kc + hi * 8);
        a1.h[1] = *(const v8h*)(arow1 + kc + 16 + hi * 8);
#pragma unroll
        for (int t = 0; t < 4; ++t) {
            // B 32x16 fragment: lanes0-15 K 0..7 / 16..23 ; lanes16-31 K 8..15 / 24..31 for column n0+t*16+mr
            const _Float16* brow = Bt + (size_t)(n0 + t * 16 + mr) * K + kc + hi * 8;
            union { v16h v; v8h h[2]; } b;
            b.h[0] = *(const v8h*)(brow);
            b.h[1] = *(const v8h*)(brow + 16);
            acc0[t] = __builtin_amdgcn_wmma_f32_16x16x32_f16(
                false, a0.v, false, b.v, (short)0, acc0[t], false, false);
            acc1[t] = __builtin_amdgcn_wmma_f32_16x16x32_f16(
                false, a1.v, false, b.v, (short)0, acc1[t], false, false);
        }
    }

#pragma unroll
    for (int t = 0; t < 4; ++t) {
        int col = n0 + t * 16 + mr;
        float bv = bias ? bias[col] : 0.f;
#pragma unroll
        for (int j = 0; j < 8; ++j) {
            int row = m0 + hi * 8 + j;          // C/D layout: VGPR j -> M = j (+8 for hi half)
            float v0 = acc0[t][j] + bv;
            float v1 = acc1[t][j] + bv;
            if (relu) { v0 = fmaxf(v0, 0.f); v1 = fmaxf(v1, 0.f); }
            size_t o0 = (size_t)row * N + col;
            size_t o1 = (size_t)(row + 16) * N + col;
            if (outF) { outF[o0] = v0; outF[o1] = v1; }
            if (outH) { outH[o0] = (_Float16)v0; outH[o1] = (_Float16)v1; }
        }
    }
}

// ---------- per-edge attention scores + segment max ----------
__global__ void edge_score_kernel(const _Float16* __restrict__ q, const _Float16* __restrict__ k,
                                  const int* __restrict__ src, const int* __restrict__ dst,
                                  float* __restrict__ scores, unsigned* __restrict__ mseg) {
    int wid = (blockIdx.x * blockDim.x + threadIdx.x) >> 5;
    int lane = threadIdx.x & 31;
    if (wid >= N_EDGES * NHEAD) return;
    int e = wid >> 3, hd = wid & 7;
    int s = src[e], d = dst[e];
    v4h qa = *(const v4h*)(q + (size_t)d * HC + hd * CHEAD + lane * 4);
    v4h ka = *(const v4h*)(k + (size_t)s * HC + hd * CHEAD + lane * 4);
    float acc = 0.f;
#pragma unroll
    for (int i = 0; i < 4; ++i) acc += (float)qa[i] * (float)ka[i];
#pragma unroll
    for (int off = 16; off >= 1; off >>= 1) acc += __shfl_xor(acc, off, 32);
    if (lane == 0) {
        float sc = acc * 0.08838834764831845f;   // 1/sqrt(128)
        scores[wid] = sc;
        atomicMax(mseg + (size_t)d * NHEAD + hd, fmap(sc));
    }
}

// ---------- ex = exp(score - max); den += ex ----------
__global__ void edge_exp_kernel(const int* __restrict__ dst, float* __restrict__ scores,
                                const unsigned* __restrict__ mseg, float* __restrict__ den) {
    int tid = blockIdx.x * blockDim.x + threadIdx.x;
    if (tid >= N_EDGES * NHEAD) return;
    int e = tid >> 3, hd = tid & 7;
    int d = dst[e];
    float ex = __expf(scores[tid] - funmap(mseg[(size_t)d * NHEAD + hd]));
    scores[tid] = ex;
    atomicAdd(den + (size_t)d * NHEAD + hd, ex);
}

// ---------- agg[dst] += (alpha/H) * v[src]  (head-mean folded in) ----------
__global__ void edge_msg_kernel(const _Float16* __restrict__ v,
                                const int* __restrict__ src, const int* __restrict__ dst,
                                const float* __restrict__ scores, const float* __restrict__ den,
                                float* __restrict__ agg) {
    int wid = (blockIdx.x * blockDim.x + threadIdx.x) >> 5;
    int lane = threadIdx.x & 31;
    if (wid >= N_EDGES * NHEAD) return;
    int e = wid >> 3, hd = wid & 7;
    int s = src[e], d = dst[e];
    float scale = scores[wid] / den[(size_t)d * NHEAD + hd] * 0.125f;  // /H
    v4h va = *(const v4h*)(v + (size_t)s * HC + hd * CHEAD + lane * 4);
    float* p = agg + (size_t)d * DMODEL + lane * 4;
#pragma unroll
    for (int i = 0; i < 4; ++i) atomicAdd(p + i, scale * (float)va[i]);
}

// ---------- h = LayerNorm(x + res) * s + b ; one wave per row ----------
__global__ void layernorm_kernel(const float* __restrict__ x, const float* __restrict__ res,
                                 const float* __restrict__ s, const float* __restrict__ b,
                                 float* __restrict__ hOut, _Float16* __restrict__ h16) {
    int wid = (blockIdx.x * blockDim.x + threadIdx.x) >> 5;
    int lane = threadIdx.x & 31;
    if (wid >= N_NODES) return;
    const float* xr = x + (size_t)wid * DMODEL;
    const float* rr = res + (size_t)wid * DMODEL;
    float v[4]; float sum = 0.f;
#pragma unroll
    for (int i = 0; i < 4; ++i) { int c = lane + 32 * i; v[i] = xr[c] + rr[c]; sum += v[i]; }
#pragma unroll
    for (int off = 16; off >= 1; off >>= 1) sum += __shfl_xor(sum, off, 32);
    float mu = sum * (1.f / 128.f);
    float var = 0.f;
#pragma unroll
    for (int i = 0; i < 4; ++i) { float dlt = v[i] - mu; var += dlt * dlt; }
#pragma unroll
    for (int off = 16; off >= 1; off >>= 1) var += __shfl_xor(var, off, 32);
    float rstd = rsqrtf(var * (1.f / 128.f) + 1e-5f);
#pragma unroll
    for (int i = 0; i < 4; ++i) {
        int c = lane + 32 * i;
        float y = (v[i] - mu) * rstd * s[c] + b[c];
        hOut[(size_t)wid * DMODEL + c] = y;
        h16[(size_t)wid * DMODEL + c] = (_Float16)y;
    }
}

// ---------- global mean pool (sum + counts via atomics) ----------
__global__ void pool_kernel(const float* __restrict__ h, const int* __restrict__ batch,
                            float* __restrict__ pooled, float* __restrict__ counts) {
    int tid = blockIdx.x * blockDim.x + threadIdx.x;
    if (tid >= N_NODES * DMODEL) return;
    int n = tid >> 7, c = tid & 127;
    int b = batch[n];
    atomicAdd(pooled + (size_t)b * DMODEL + c, h[tid]);
    if (c == 0) atomicAdd(counts + b, 1.f);
}

// ---------- classifier: relu(pooled/cnt @ w1 + b1) @ w2 + b2 ----------
__global__ void classifier_kernel(const float* __restrict__ pooled, const float* __restrict__ counts,
                                  const float* __restrict__ w1, const float* __restrict__ b1,
                                  const float* __restrict__ w2, const float* __restrict__ b2,
                                  float* __restrict__ out) {
    __shared__ float hid[64];
    int bg = blockIdx.x;
    float inv = 1.f / fmaxf(counts[bg], 1.f);
    int j = threadIdx.x;
    if (j < 64) {
        float a = b1[j];
        for (int c = 0; c < DMODEL; ++c) a += pooled[(size_t)bg * DMODEL + c] * inv * w1[c * 64 + j];
        hid[j] = fmaxf(a, 0.f);
    }
    __syncthreads();
    if (j < 10) {
        float a = b2[j];
        for (int c = 0; c < 64; ++c) a += hid[c] * w2[c * 10 + j];
        out[bg * 10 + j] = a;
    }
}

extern "C" void kernel_launch(void* const* d_in, const int* in_sizes, int n_in,
                              void* d_out, int out_size, void* d_ws, size_t ws_size,
                              hipStream_t stream) {
    (void)in_sizes; (void)n_in; (void)out_size; (void)ws_size;
    const float* x     = (const float*)d_in[0];
    const int*   ei    = (const int*)d_in[1];
    const int*   batch = (const int*)d_in[2];
    const int*   ts    = (const int*)d_in[3];
    const float* in_w  = (const float*)d_in[4];
    const float* in_b  = (const float*)d_in[5];
    const float* Wq    = (const float*)d_in[6];
    const float* bq    = (const float*)d_in[7];
    const float* Wk    = (const float*)d_in[8];
    const float* bk    = (const float*)d_in[9];
    const float* Wv    = (const float*)d_in[10];
    const float* bv    = (const float*)d_in[11];
    const float* Wskip = (const float*)d_in[12];
    const float* bskip = (const float*)d_in[13];
    const float* n1_s  = (const float*)d_in[14];
    const float* n1_b  = (const float*)d_in[15];
    const float* f_w1  = (const float*)d_in[16];
    const float* f_b1  = (const float*)d_in[17];
    const float* f_w2  = (const float*)d_in[18];
    const float* f_b2  = (const float*)d_in[19];
    const float* n2_s  = (const float*)d_in[20];
    const float* n2_b  = (const float*)d_in[21];
    const float* c_w1  = (const float*)d_in[22];
    const float* c_b1  = (const float*)d_in[23];
    const float* c_w2  = (const float*)d_in[24];
    const float* c_b2  = (const float*)d_in[25];
    const int* src  = ei;
    const int* dstE = ei + N_EDGES;

    // ---- workspace carve ----
    char* ws = (char*)d_ws;
    size_t off = 0;
    auto alloc = [&](size_t bytes) -> char* {
        char* p = ws + off; off += (bytes + 255) & ~(size_t)255; return p;
    };
    float*    h      = (float*)alloc((size_t)N_NODES * DMODEL * 4);
    _Float16* h16    = (_Float16*)alloc((size_t)N_NODES * DMODEL * 2);
    _Float16* q16    = (_Float16*)alloc((size_t)N_NODES * HC * 2);
    _Float16* k16    = (_Float16*)alloc((size_t)N_NODES * HC * 2);
    _Float16* v16    = (_Float16*)alloc((size_t)N_NODES * HC * 2);
    _Float16* ff1h   = (_Float16*)alloc((size_t)N_NODES * FFDIM * 2);
    float*    tmpND  = (float*)alloc((size_t)N_NODES * DMODEL * 4);  // agg, then ffn2 out
    float*    scores = (float*)alloc((size_t)N_EDGES * NHEAD * 4);
    unsigned* mseg   = (unsigned*)alloc((size_t)N_NODES * NHEAD * 4);
    float*    den    = (float*)alloc((size_t)N_NODES * NHEAD * 4);
    float*    pooled = (float*)alloc((size_t)BGRAPH * DMODEL * 4);
    float*    counts = (float*)alloc((size_t)BGRAPH * 4);
    _Float16* wq_t   = (_Float16*)alloc((size_t)NLAYER * HC * DMODEL * 2);
    _Float16* wk_t   = (_Float16*)alloc((size_t)NLAYER * HC * DMODEL * 2);
    _Float16* wv_t   = (_Float16*)alloc((size_t)NLAYER * HC * DMODEL * 2);
    _Float16* wsk_t  = (_Float16*)alloc((size_t)NLAYER * DMODEL * DMODEL * 2);
    _Float16* w1_t   = (_Float16*)alloc((size_t)NLAYER * FFDIM * DMODEL * 2);
    _Float16* w2_t   = (_Float16*)alloc((size_t)NLAYER * DMODEL * FFDIM * 2);

    auto pack = [&](const float* W, _Float16* Wt, int K, int N) {
        int elems = K * N;
        pack_wt_kernel<<<(elems + 255) / 256, 256, 0, stream>>>(W, Wt, K, N);
    };
    auto gemm = [&](const _Float16* A, const _Float16* Bt, const float* bias,
                    float* oF, _Float16* oH, int M, int K, int N, int relu) {
        dim3 grid(M / 32, (N + 255) / 256);
        wmma_gemm_kernel<<<grid, 128, 0, stream>>>(A, Bt, bias, oF, oH, M, K, N, relu);
    };

    // pack all weights to f16 [N][K]
    for (int l = 0; l < NLAYER; ++l) {
        pack(Wq + (size_t)l * DMODEL * HC,     wq_t + (size_t)l * HC * DMODEL,    DMODEL, HC);
        pack(Wk + (size_t)l * DMODEL * HC,     wk_t + (size_t)l * HC * DMODEL,    DMODEL, HC);
        pack(Wv + (size_t)l * DMODEL * HC,     wv_t + (size_t)l * HC * DMODEL,    DMODEL, HC);
        pack(Wskip + (size_t)l * DMODEL * DMODEL, wsk_t + (size_t)l * DMODEL * DMODEL, DMODEL, DMODEL);
        pack(f_w1 + (size_t)l * DMODEL * FFDIM, w1_t + (size_t)l * FFDIM * DMODEL, DMODEL, FFDIM);
        pack(f_w2 + (size_t)l * FFDIM * DMODEL, w2_t + (size_t)l * DMODEL * FFDIM, FFDIM, DMODEL);
    }

    // input projection + positional encoding
    init_h_kernel<<<(N_NODES * DMODEL + 255) / 256, 256, 0, stream>>>(x, ts, in_w, in_b, h, h16);

    const int edgeWaveThreads = N_EDGES * NHEAD * 32;
    for (int l = 0; l < NLAYER; ++l) {
        const _Float16* wq  = wq_t  + (size_t)l * HC * DMODEL;
        const _Float16* wk  = wk_t  + (size_t)l * HC * DMODEL;
        const _Float16* wv  = wv_t  + (size_t)l * HC * DMODEL;
        const _Float16* wsk = wsk_t + (size_t)l * DMODEL * DMODEL;
        const _Float16* w1  = w1_t  + (size_t)l * FFDIM * DMODEL;
        const _Float16* w2  = w2_t  + (size_t)l * DMODEL * FFDIM;

        // QKV projections (f16 out) + skip GEMM (f32 -> tmpND, includes bias)
        gemm(h16, wq, bq + (size_t)l * HC, nullptr, q16, N_NODES, DMODEL, HC, 0);
        gemm(h16, wk, bk + (size_t)l * HC, nullptr, k16, N_NODES, DMODEL, HC, 0);
        gemm(h16, wv, bv + (size_t)l * HC, nullptr, v16, N_NODES, DMODEL, HC, 0);
        gemm(h16, wsk, bskip + (size_t)l * DMODEL, tmpND, nullptr, N_NODES, DMODEL, DMODEL, 0);

        // segment softmax over incoming edges + message scatter (head-mean folded)
        hipMemsetAsync(mseg, 0, (size_t)N_NODES * NHEAD * 4, stream);
        hipMemsetAsync(den,  0, (size_t)N_NODES * NHEAD * 4, stream);
        edge_score_kernel<<<(edgeWaveThreads + 255) / 256, 256, 0, stream>>>(q16, k16, src, dstE, scores, mseg);
        edge_exp_kernel<<<(N_EDGES * NHEAD + 255) / 256, 256, 0, stream>>>(dstE, scores, mseg, den);
        edge_msg_kernel<<<(edgeWaveThreads + 255) / 256, 256, 0, stream>>>(v16, src, dstE, scores, den, tmpND);

        // h = LN(attn_out + h)
        layernorm_kernel<<<(N_NODES * 32 + 255) / 256, 256, 0, stream>>>(
            tmpND, h, n1_s + (size_t)l * DMODEL, n1_b + (size_t)l * DMODEL, h, h16);

        // FFN: relu(h@w1+b1)@w2+b2, then h = LN(ff + h)
        gemm(h16, w1, f_b1 + (size_t)l * FFDIM, nullptr, ff1h, N_NODES, DMODEL, FFDIM, 1);
        gemm(ff1h, w2, f_b2 + (size_t)l * DMODEL, tmpND, nullptr, N_NODES, FFDIM, DMODEL, 0);
        layernorm_kernel<<<(N_NODES * 32 + 255) / 256, 256, 0, stream>>>(
            tmpND, h, n2_s + (size_t)l * DMODEL, n2_b + (size_t)l * DMODEL, h, h16);
    }

    // global mean pool + classifier
    hipMemsetAsync(pooled, 0, (size_t)BGRAPH * DMODEL * 4, stream);
    hipMemsetAsync(counts, 0, (size_t)BGRAPH * 4, stream);
    pool_kernel<<<(N_NODES * DMODEL + 255) / 256, 256, 0, stream>>>(h, batch, pooled, counts);
    classifier_kernel<<<BGRAPH, 64, 0, stream>>>(pooled, counts, c_w1, c_b1, c_w2, c_b2, (float*)d_out);
}
